// GatingNetwork_16664473108708
// MI455X (gfx1250) — matrix operand order repaired and checked by reading, not compiled, not run on hardware
//
#include <hip/hip_runtime.h>
#include <hip/hip_bf16.h>
#include <math.h>

typedef __attribute__((ext_vector_type(16))) _Float16 v16h;
typedef __attribute__((ext_vector_type(8)))  float    v8f;

#define B_    4096
#define E_    8
#define C_    1000
#define D_    8029
#define DP_   8192          // D padded to 256 K-tiles (branch-free async pipeline)
#define KT1   (DP_/32)      // 256 K-tiles for GEMM1
#define CHUNK 8             // K-tiles staged per async LDS copy
#define NCH   (KT1/CHUNK)   // 32 chunks
#define H1_   256
#define H2_   128
#define KT2   (H1_/32)      // 8 K-tiles for GEMM2
#define EPS_    1e-8f
#define LN_EPS_ 1e-5f

// ---------------------------------------------------------------------------
// WMMA fragment index helpers (ISA 7.12.2, 16-bit layouts, wave32)
// A 16x32 f16: lane = m + ((kk>>3)&1)*16 ; half = (kk&7) + (kk>=16 ? 8 : 0)
// B 32x16 f16: lane = n + ((kk>>3)&1)*16 ; half = (kk&7) + (kk>=16 ? 8 : 0)
// Stored as [tile][kt][lane][half] so each lane loads 16 contiguous halves.
// ---------------------------------------------------------------------------
__device__ __forceinline__ size_t aFragIdx(int tile, int KT, int kt, int m, int kk) {
  int mid = (kk >> 3) & 1;
  int hi  = (kk >> 4) & 1;
  return ((size_t)(tile * KT + kt) * 32 + (m + (mid << 4))) * 16 + ((kk & 7) + (hi << 3));
}
__device__ __forceinline__ size_t bFragIdx(int tile, int KT, int kt, int n, int kk) {
  int mid = (kk >> 3) & 1;
  int hi  = (kk >> 4) & 1;
  return ((size_t)(tile * KT + kt) * 32 + (n + (mid << 4))) * 16 + ((kk & 7) + (hi << 3));
}

// ---------------------------------------------------------------------------
// Weight packing: f32 row-major -> f16 B-fragment layout (K padded w/ zeros)
// ---------------------------------------------------------------------------
__global__ void pack_w1(const float* __restrict__ W1, _Float16* __restrict__ w1P) {
  int idx = blockIdx.x * 256 + threadIdx.x;          // DP_*H1_ elements
  if (idx >= DP_ * H1_) return;
  int k = idx >> 8;                                  // 0..8191
  int n = idx & 255;
  float v = (k < D_) ? W1[(size_t)k * H1_ + n] : 0.f;
  w1P[bFragIdx(n >> 4, KT1, k >> 5, n & 15, k & 31)] = (_Float16)v;
}

__global__ void pack_w2(const float* __restrict__ W2, _Float16* __restrict__ w2P) {
  int idx = blockIdx.x * 256 + threadIdx.x;          // 256*128 elements
  if (idx >= H1_ * H2_) return;
  int k = idx >> 7;                                  // 0..255
  int n = idx & 127;
  float v = W2[(size_t)k * H2_ + n];
  w2P[bFragIdx(n >> 4, KT2, k >> 5, n & 15, k & 31)] = (_Float16)v;
}

// ---------------------------------------------------------------------------
// Feature kernel: one 256-thread block per batch row.
// Computes the 8029 features and writes them f16 A-fragment-packed.
// ---------------------------------------------------------------------------
__global__ __launch_bounds__(256) void feat_kernel(const float* __restrict__ post,
                                                   _Float16* __restrict__ featsP) {
  extern __shared__ float sm[];
  float* sp     = sm;                 // 8000 : p
  float* slp    = sp + 8000;          // 8000 : log(p+eps)
  float* scross = slp + 8000;         // 64   : cross[i][j] = sum p_i * logp_j
  float* sred   = scross + 64;        // 256  : reduction scratch
  float* sent   = sred + 256;         // 8    : entropy
  float* sconf  = sent + 8;           // 8    : confidence
  float* smarg  = sconf + 8;          // 8    : margin
  float* ssc    = smarg + 8;          // 8    : [disag, mean_kl, mixH, pvar, mi]
  int*   stop1  = (int*)(ssc + 8);    // 8    : argmax class per expert

  const int b    = blockIdx.x;
  const int tid  = threadIdx.x;
  const int lane = tid & 31;
  const int wid  = tid >> 5;
  const float* p = post + (size_t)b * (E_ * C_);

  for (int i = tid; i < E_ * C_; i += 256) {
    float v = p[i];
    sp[i]  = v;
    slp[i] = logf(v + EPS_);
  }
  __syncthreads();

  // cross[i][j]: 64 pairs, 4 lanes per pair
  {
    int pairId = tid >> 2, sub = tid & 3;
    int i = pairId >> 3, j = pairId & 7;
    float acc = 0.f;
    for (int c = sub; c < C_; c += 4)
      acc += sp[i * C_ + c] * slp[j * C_ + c];
    acc += __shfl_xor(acc, 1, 32);
    acc += __shfl_xor(acc, 2, 32);
    if (sub == 0) scross[pairId] = acc;
  }

  // per-expert max / top2 / argmax: wave `wid` handles expert `wid`
  {
    float m1 = -1.f, m2 = -1.f; int am = 0;
    for (int c = lane; c < C_; c += 32) {
      float v = sp[wid * C_ + c];
      if (v > m1)      { m2 = m1; m1 = v; am = c; }
      else if (v > m2) { m2 = v; }
    }
    for (int d = 16; d >= 1; d >>= 1) {
      float o1 = __shfl_xor(m1, d, 32);
      float o2 = __shfl_xor(m2, d, 32);
      int   oa = __shfl_xor(am, d, 32);
      if (o1 > m1)       { m2 = fmaxf(m1, o2); m1 = o1; am = oa; }
      else if (o1 == m1) { m2 = fmaxf(m2, fmaxf(o1, o2)); am = (oa < am) ? oa : am; }
      else               { m2 = fmaxf(m2, o1); }
    }
    if (lane == 0) {
      sconf[wid] = m1;
      smarg[wid] = m1 - m2;
      stop1[wid] = am;
    }
  }
  __syncthreads();

  if (tid < E_) sent[tid] = -scross[tid * 9];   // entropy = -plogp (diagonal)

  // mix entropy + posterior variance (ddof=1), thread-strided over classes
  float meAcc = 0.f, pvAcc = 0.f;
  for (int c = tid; c < C_; c += 256) {
    float vals[E_]; float s = 0.f;
    #pragma unroll
    for (int i = 0; i < E_; ++i) { vals[i] = sp[i * C_ + c]; s += vals[i]; }
    float mean = s * 0.125f;
    meAcc -= mean * logf(mean + EPS_);
    float ss = 0.f;
    #pragma unroll
    for (int i = 0; i < E_; ++i) { float d2 = vals[i] - mean; ss += d2 * d2; }
    pvAcc += ss * (1.0f / 7.0f);
  }
  sred[tid] = meAcc; __syncthreads();
  for (int s2 = 128; s2 > 0; s2 >>= 1) { if (tid < s2) sred[tid] += sred[tid + s2]; __syncthreads(); }
  if (tid == 0) ssc[2] = sred[0];
  __syncthreads();
  sred[tid] = pvAcc; __syncthreads();
  for (int s2 = 128; s2 > 0; s2 >>= 1) { if (tid < s2) sred[tid] += sred[tid + s2]; __syncthreads(); }

  if (tid == 0) {
    ssc[3] = sred[0] * (1.0f / C_);
    float kl = 0.f;
    for (int i = 0; i < E_; ++i)
      for (int j = i + 1; j < E_; ++j)
        kl += scross[i * 8 + i] - scross[i * 8 + j];
    ssc[1] = kl * (1.0f / 28.0f);
    int t[E_];
    for (int i = 0; i < E_; ++i) t[i] = stop1[i];
    for (int i = 1; i < E_; ++i) {                 // insertion sort
      int v = t[i], j = i - 1;
      while (j >= 0 && t[j] > v) { t[j + 1] = t[j]; --j; }
      t[j + 1] = v;
    }
    int nu = 1;
    for (int i = 1; i < E_; ++i) nu += (t[i] != t[i - 1]);
    ssc[0] = (float)(nu - 1) * (1.0f / 7.0f);
    float esum = 0.f;
    for (int i = 0; i < E_; ++i) esum += sent[i];
    ssc[4] = ssc[2] - esum * 0.125f;               // mutual info
  }
  __syncthreads();

  // write feats, A-fragment-packed f16, zero padded to 8192
  const int tileM = b >> 4, m = b & 15;
  for (int d = tid; d < DP_; d += 256) {
    float v;
    if      (d < 8000) v = sp[d];
    else if (d < 8008) v = sent[d - 8000];
    else if (d < 8016) v = sconf[d - 8008];
    else if (d < 8024) v = smarg[d - 8016];
    else if (d < 8029) v = ssc[d - 8024];
    else               v = 0.f;
    featsP[aFragIdx(tileM, KT1, d >> 5, m, d & 31)] = (_Float16)v;
  }
}

// ---------------------------------------------------------------------------
// GEMM1: h1 = relu(LN(feats @ W1 + b1)); output A-fragment-packed f16.
// 256 blocks (16 rows each) x 8 waves; wave w -> N-tiles 2w, 2w+1.
// A tile staged into LDS with double-buffered global_load_async_to_lds_b128;
// B streams straight from global into WMMA operands.
// ---------------------------------------------------------------------------
__global__ __launch_bounds__(256) void gemm1(const _Float16* __restrict__ featsP,
                                             const _Float16* __restrict__ w1P,
                                             const float* __restrict__ b1,
                                             const float* __restrict__ g1,
                                             const float* __restrict__ be1,
                                             _Float16* __restrict__ h1P) {
  __shared__ __align__(16) _Float16 sA[2][CHUNK * 512];  // 8 KB per buffer
  __shared__ float sC[16][H1_];                          // 16 KB
  const int tileM = blockIdx.x;
  const int tid = threadIdx.x, lane = tid & 31, w = tid >> 5;

  const _Float16* Abase = featsP + (size_t)tileM * KT1 * 512;   // halves

  // async copy of one CHUNK (8 KB) into LDS buffer `buf`: 256 thr x 2 x 16B
  auto issue = [&](int c, int buf) {
    const _Float16* src = Abase + (size_t)c * (CHUNK * 512);
    #pragma unroll
    for (int i = 0; i < 2; ++i) {
      int off = (tid + i * 256) * 8;                   // halves (16 B units)
      unsigned ldsAddr = (unsigned)(uintptr_t)(void*)&sA[buf][off]; // addr[31:0] = LDS offset
      const _Float16* g = src + off;
      asm volatile("global_load_async_to_lds_b128 %0, %1, off"
                   :: "v"(ldsAddr), "v"(g) : "memory");
    }
  };

  issue(0, 0);

  v8f acc0 = {}, acc1 = {};
  const v16h* B0 = (const v16h*)w1P + (size_t)(2 * w)     * KT1 * 32 + lane;
  const v16h* B1 = (const v16h*)w1P + (size_t)(2 * w + 1) * KT1 * 32 + lane;

  for (int c = 0; c < NCH; ++c) {
    asm volatile("s_wait_asynccnt 0x0" ::: "memory");  // my copies for buf[c&1] landed
    __syncthreads();                                   // everyone's landed; prev compute done
    if (c + 1 < NCH) issue(c + 1, (c + 1) & 1);

    const v16h* Al = (const v16h*)&sA[c & 1][0] + lane;
    #pragma unroll
    for (int j = 0; j < CHUNK; ++j) {
      int kt = c * CHUNK + j;
      v16h a   = Al[j * 32];                           // ds_load_b128 x2 (shared A)
      v16h bb0 = B0[kt * 32];
      v16h bb1 = B1[kt * 32];
      acc0 = __builtin_amdgcn_wmma_f32_16x16x32_f16(false, a, false, bb0, (short)0, acc0, false, false);
      acc1 = __builtin_amdgcn_wmma_f32_16x16x32_f16(false, a, false, bb1, (short)0, acc1, false, false);
    }
  }

  const int n0 = 2 * w, n1 = 2 * w + 1;
  const int nlo = lane & 15, rowAdd = (lane >> 4) << 3;
  #pragma unroll
  for (int r = 0; r < 8; ++r) {
    int row = r + rowAdd;
    int nA = n0 * 16 + nlo, nB = n1 * 16 + nlo;
    sC[row][nA] = acc0[r] + b1[nA];
    sC[row][nB] = acc1[r] + b1[nB];
  }
  __syncthreads();

  // LayerNorm(256) + ReLU; wave w handles rows 2w, 2w+1
  #pragma unroll
  for (int rr = 0; rr < 2; ++rr) {
    int row = 2 * w + rr;
    float x[8]; float s = 0.f;
    #pragma unroll
    for (int j = 0; j < 8; ++j) { x[j] = sC[row][lane + 32 * j]; s += x[j]; }
    for (int d = 16; d; d >>= 1) s += __shfl_xor(s, d, 32);
    float mu = s * (1.0f / H1_);
    float vs = 0.f;
    #pragma unroll
    for (int j = 0; j < 8; ++j) { float dd = x[j] - mu; vs += dd * dd; }
    for (int d = 16; d; d >>= 1) vs += __shfl_xor(vs, d, 32);
    float rs = rsqrtf(vs * (1.0f / H1_) + LN_EPS_);
    #pragma unroll
    for (int j = 0; j < 8; ++j) {
      int n = lane + 32 * j;                     // kt = j, kk = lane
      float y = fmaxf((x[j] - mu) * rs * g1[n] + be1[n], 0.f);
      h1P[aFragIdx(tileM, KT2, j, row, lane)] = (_Float16)y;
    }
  }
}

// ---------------------------------------------------------------------------
// GEMM2 + LN + ReLU + tiny GEMM3 + softmax, fully fused per 16-row block.
// ---------------------------------------------------------------------------
__global__ __launch_bounds__(256) void gemm2_tail(const _Float16* __restrict__ h1P,
                                                  const _Float16* __restrict__ w2P,
                                                  const float* __restrict__ b2,
                                                  const float* __restrict__ g2,
                                                  const float* __restrict__ be2,
                                                  const float* __restrict__ W3,
                                                  const float* __restrict__ b3,
                                                  float* __restrict__ out) {
  __shared__ float sH[16][H2_];
  __shared__ float sR[16][H2_];
  const int tileM = blockIdx.x;
  const int tid = threadIdx.x, lane = tid & 31, w = tid >> 5;

  v8f acc = {};
  const v16h* A  = (const v16h*)h1P + (size_t)tileM * KT2 * 32 + lane;
  const v16h* Bp = (const v16h*)w2P + (size_t)w     * KT2 * 32 + lane;
  #pragma unroll
  for (int kt = 0; kt < KT2; ++kt) {
    v16h a  = A [kt * 32];
    v16h bb = Bp[kt * 32];
    acc = __builtin_amdgcn_wmma_f32_16x16x32_f16(false, a, false, bb, (short)0, acc, false, false);
  }
  const int nlo = lane & 15, rowAdd = (lane >> 4) << 3;
  #pragma unroll
  for (int r = 0; r < 8; ++r) {
    int n = w * 16 + nlo;
    sH[r + rowAdd][n] = acc[r] + b2[n];
  }
  __syncthreads();

  // LayerNorm(128) + ReLU; wave w handles rows 2w, 2w+1
  #pragma unroll
  for (int rr = 0; rr < 2; ++rr) {
    int row = 2 * w + rr;
    float x[4]; float s = 0.f;
    #pragma unroll
    for (int j = 0; j < 4; ++j) { x[j] = sH[row][lane + 32 * j]; s += x[j]; }
    for (int d = 16; d; d >>= 1) s += __shfl_xor(s, d, 32);
    float mu = s * (1.0f / H2_);
    float vs = 0.f;
    #pragma unroll
    for (int j = 0; j < 4; ++j) { float dd = x[j] - mu; vs += dd * dd; }
    for (int d = 16; d; d >>= 1) vs += __shfl_xor(vs, d, 32);
    float rs = rsqrtf(vs * (1.0f / H2_) + LN_EPS_);
    #pragma unroll
    for (int j = 0; j < 4; ++j) {
      int n = lane + 32 * j;
      sR[row][n] = fmaxf((x[j] - mu) * rs * g2[n] + be2[n], 0.f);
    }
  }
  __syncthreads();

  // logits = sR @ W3 + b3; softmax over 8 experts via 8-lane shuffle groups
  if (tid < 128) {
    int r = tid >> 3, e = tid & 7;
    float l = b3[e];
    for (int k = 0; k < H2_; ++k) l += sR[r][k] * W3[k * E_ + e];
    float mx = l;
    for (int d = 4; d; d >>= 1) mx = fmaxf(mx, __shfl_xor(mx, d, 32));
    float ex = expf(l - mx);
    float sm = ex;
    for (int d = 4; d; d >>= 1) sm += __shfl_xor(sm, d, 32);
    out[((size_t)tileM * 16 + r) * E_ + e] = ex / sm;
  }
}

// ---------------------------------------------------------------------------
extern "C" void kernel_launch(void* const* d_in, const int* in_sizes, int n_in,
                              void* d_out, int out_size, void* d_ws, size_t ws_size,
                              hipStream_t stream) {
  const float* post = (const float*)d_in[0];
  const float* W1   = (const float*)d_in[1];
  const float* b1   = (const float*)d_in[2];
  const float* g1   = (const float*)d_in[3];
  const float* be1  = (const float*)d_in[4];
  const float* W2   = (const float*)d_in[5];
  const float* b2   = (const float*)d_in[6];
  const float* g2   = (const float*)d_in[7];
  const float* be2  = (const float*)d_in[8];
  const float* W3   = (const float*)d_in[9];
  const float* b3   = (const float*)d_in[10];
  float* out = (float*)d_out;

  // workspace layout (all offsets 256B-aligned)
  char* ws = (char*)d_ws;
  const size_t W1P_BYTES   = (size_t)DP_ * H1_ * 2;       //  4,194,304
  const size_t W2P_BYTES   = (size_t)H1_ * H2_ * 2;       //     65,536
  const size_t FEATS_BYTES = (size_t)B_ * DP_ * 2;        // 67,108,864
  _Float16* w1P    = (_Float16*)(ws);
  _Float16* w2P    = (_Float16*)(ws + W1P_BYTES);
  _Float16* featsP = (_Float16*)(ws + W1P_BYTES + W2P_BYTES);
  _Float16* h1P    = (_Float16*)(ws + W1P_BYTES + W2P_BYTES + FEATS_BYTES);

  pack_w1<<<(DP_ * H1_ + 255) / 256, 256, 0, stream>>>(W1, w1P);
  pack_w2<<<(H1_ * H2_ + 255) / 256, 256, 0, stream>>>(W2, w2P);

  const size_t featSmem = (8000 + 8000 + 64 + 256 + 8 + 8 + 8 + 8) * sizeof(float)
                        + 8 * sizeof(int);                // ~65.5 KB dynamic LDS
  feat_kernel<<<B_, 256, featSmem, stream>>>(post, featsP);

  gemm1<<<B_ / 16, 256, 0, stream>>>(featsP, w1P, b1, g1, be1, h1P);
  gemm2_tail<<<B_ / 16, 256, 0, stream>>>(h1P, w2P, b2, g2, be2, W3, b3, out);
}
